// SelfAttention_90099823936171
// MI455X (gfx1250) — compile-verified, matrix operand order checked by reading, not run on hardware
//
#include <hip/hip_runtime.h>

typedef __bf16 bf16_t;
typedef __attribute__((ext_vector_type(4)))  __bf16 bf16x4;
typedef __attribute__((ext_vector_type(8)))  __bf16 bf16x8;
typedef __attribute__((ext_vector_type(16))) __bf16 bf16x16;
typedef __attribute__((ext_vector_type(8)))  float  f32x8;

#define B_  2
#define S_  2048
#define H_  16
#define DH_ 64
#define D_  1024
#define M_  (B_ * S_)        // 4096 rows of x

// ---------------------------------------------------------------------------
// Pass 0a: fp32 -> bf16 conversion of x (row-major [M_, D_])
// ---------------------------------------------------------------------------
__global__ void cvt_x_kernel(const float* __restrict__ x, bf16_t* __restrict__ xb) {
    size_t i = ((size_t)blockIdx.x * blockDim.x + threadIdx.x) * 4;
    if (i >= (size_t)M_ * D_) return;
    float4 f = *(const float4*)(x + i);
    bf16x4 o;
    o[0] = (bf16_t)f.x; o[1] = (bf16_t)f.y; o[2] = (bf16_t)f.z; o[3] = (bf16_t)f.w;
    *(bf16x4*)(xb + i) = o;
}

// ---------------------------------------------------------------------------
// Pass 0b: W [K][N] fp32 -> Wt [N][K] bf16 (tiled LDS transpose)
// ---------------------------------------------------------------------------
__global__ void transpose_w_kernel(const float* __restrict__ W, bf16_t* __restrict__ Wt) {
    __shared__ float tile[32][33];
    int k0 = blockIdx.x * 32;
    int n0 = blockIdx.y * 32;
    int tx = threadIdx.x;          // 0..31
    int ty = threadIdx.y;          // 0..7
#pragma unroll
    for (int r = 0; r < 32; r += 8)
        tile[ty + r][tx] = W[(size_t)(k0 + ty + r) * D_ + (n0 + tx)];
    __syncthreads();
#pragma unroll
    for (int r = 0; r < 32; r += 8)
        Wt[(size_t)(n0 + ty + r) * D_ + (k0 + tx)] = (bf16_t)tile[tx][ty + r];
}

// ---------------------------------------------------------------------------
// Pass 1: QKV projection GEMM. One wave computes a 64x64 output tile as a
// 4x4 grid of 16x16 WMMA sub-tiles: 16 wmma per 32-K step from 16 b128 loads.
//   out Q,K : bf16 [B,H,S,DH]
//   out Vt  : bf16 [B,H,DH,S]   (transposed for PV B-fragment loads)
// ---------------------------------------------------------------------------
__global__ void __launch_bounds__(256)
qkv_gemm_kernel(const bf16_t* __restrict__ xb,
                const bf16_t* __restrict__ wtq, const bf16_t* __restrict__ wtk,
                const bf16_t* __restrict__ wtv,
                const float* __restrict__ bq, const float* __restrict__ bk,
                const float* __restrict__ bv,
                bf16_t* __restrict__ qo, bf16_t* __restrict__ ko,
                bf16_t* __restrict__ vto) {
    int wave = blockIdx.x * (blockDim.x >> 5) + (threadIdx.x >> 5);
    int lane = threadIdx.x & 31;
    int row  = lane & 15;      // A row / B col / C col within a 16x16 sub-tile
    int half = lane >> 4;      // K-chunk select

    int mat = wave >> 10;                   // 1024 tiles per matrix; 0=Q 1=K 2=V
    int rem = wave & 1023;
    int m0  = (rem >> 4) * 64;              // 64 M-tiles
    int n0  = (rem & 15) * 64;              // 16 N-tiles

    const bf16_t* wt   = (mat == 0) ? wtq : ((mat == 1) ? wtk : wtv);
    const float*  bias = (mat == 0) ? bq  : ((mat == 1) ? bk  : bv);

    const bf16_t* aBase = xb + (size_t)(m0 + row) * D_ + 8 * half;
    const bf16_t* bBase = wt + (size_t)(n0 + row) * D_ + 8 * half;

    f32x8 acc[4][4] = {};

    for (int kk = 0; kk < D_; kk += 32) {
        bf16x16 A[4], Bf[4];
#pragma unroll
        for (int mi = 0; mi < 4; ++mi) {
            bf16x8 lo = *(const bf16x8*)(aBase + (size_t)mi * 16 * D_ + kk);
            bf16x8 hi = *(const bf16x8*)(aBase + (size_t)mi * 16 * D_ + kk + 16);
#pragma unroll
            for (int i = 0; i < 8; ++i) { A[mi][i] = lo[i]; A[mi][8 + i] = hi[i]; }
        }
#pragma unroll
        for (int ni = 0; ni < 4; ++ni) {
            bf16x8 lo = *(const bf16x8*)(bBase + (size_t)ni * 16 * D_ + kk);
            bf16x8 hi = *(const bf16x8*)(bBase + (size_t)ni * 16 * D_ + kk + 16);
#pragma unroll
            for (int i = 0; i < 8; ++i) { Bf[ni][i] = lo[i]; Bf[ni][8 + i] = hi[i]; }
        }
#pragma unroll
        for (int mi = 0; mi < 4; ++mi)
#pragma unroll
            for (int ni = 0; ni < 4; ++ni)
                acc[mi][ni] = __builtin_amdgcn_wmma_f32_16x16x32_bf16(
                    false, A[mi], false, Bf[ni], (short)0, acc[mi][ni], false, false);
    }

    // ---- epilogue: bias add + bf16 stores ----
    int h     = n0 >> 6;                    // 64-wide tile spans exactly one head
    int b     = m0 >> 11;                   // S_ = 2048, tile fits in one batch
    int sBase = m0 & (S_ - 1);

    float bcol[4];
#pragma unroll
    for (int ni = 0; ni < 4; ++ni) bcol[ni] = bias[n0 + ni * 16 + row];

    if (mat == 2) {
#pragma unroll
        for (int mi = 0; mi < 4; ++mi)
#pragma unroll
            for (int ni = 0; ni < 4; ++ni) {
                int dh = ni * 16 + row;
                bf16x8 pack;
#pragma unroll
                for (int v = 0; v < 8; ++v) pack[v] = (bf16_t)(acc[mi][ni][v] + bcol[ni]);
                size_t base = (((size_t)b * H_ + h) * DH_ + dh) * S_
                              + sBase + mi * 16 + 8 * half;
                *(bf16x8*)(vto + base) = pack;
            }
    } else {
        bf16_t* o = (mat == 0) ? qo : ko;
#pragma unroll
        for (int mi = 0; mi < 4; ++mi)
#pragma unroll
            for (int ni = 0; ni < 4; ++ni) {
                int dh = ni * 16 + row;
                size_t base = (((size_t)b * H_ + h) * S_ + sBase + mi * 16) * DH_ + dh;
#pragma unroll
                for (int v = 0; v < 8; ++v)
                    o[base + (size_t)(v + 8 * half) * DH_] =
                        (bf16_t)(acc[mi][ni][v] + bcol[ni]);
            }
    }
}

// ---------------------------------------------------------------------------
// Pass 2: causal flash attention. One wave per 32-query tile (two 16-row
// sub-tiles sharing K/V fragments); 32 keys per iteration -> 16 wmma : 16 loads.
// ---------------------------------------------------------------------------
__global__ void __launch_bounds__(128)
attn_kernel(const bf16_t* __restrict__ q, const bf16_t* __restrict__ k,
            const bf16_t* __restrict__ vt, float* __restrict__ out) {
    __shared__ bf16_t plds[4][2 * 16 * 32];   // per-wave P staging (C->A layout)
    int wid  = threadIdx.x >> 5;
    int lane = threadIdx.x & 31;
    int wave = blockIdx.x * 4 + wid;

    int qt = wave & 63;                        // S_/32 = 64 query tiles
    int h  = (wave >> 6) & (H_ - 1);
    int b  = wave >> 10;
    int row  = lane & 15;
    int half = lane >> 4;

    const bf16_t* qh = q  + (((size_t)b * H_ + h) * S_) * DH_;
    const bf16_t* kh = k  + (((size_t)b * H_ + h) * S_) * DH_;
    const bf16_t* vh = vt + (((size_t)b * H_ + h) * DH_) * S_;

    int q0 = qt * 32;

    // Q A-fragments: [sub-tile mi][dh chunk f: 0..31 / 32..63]
    bf16x16 QA[2][2];
#pragma unroll
    for (int mi = 0; mi < 2; ++mi) {
        const bf16_t* qr = qh + (size_t)(q0 + mi * 16 + row) * DH_;
#pragma unroll
        for (int f = 0; f < 2; ++f) {
            bf16x8 lo = *(const bf16x8*)(qr + f * 32 + 8 * half);
            bf16x8 hi = *(const bf16x8*)(qr + f * 32 + 16 + 8 * half);
#pragma unroll
            for (int i = 0; i < 8; ++i) { QA[mi][f][i] = lo[i]; QA[mi][f][8 + i] = hi[i]; }
        }
    }

    f32x8 acc[2][4] = {};
    float mrow[2][8], lrow[2][8];
#pragma unroll
    for (int mi = 0; mi < 2; ++mi)
#pragma unroll
        for (int v = 0; v < 8; ++v) { mrow[mi][v] = -3.0e38f; lrow[mi][v] = 0.0f; }

    int nIter = qt + 1;                        // exact causal bound for 32-blocks
    bf16_t* pl = plds[wid];

    for (int jj = 0; jj < nIter; ++jj) {
        int kb = jj * 32;

        // ---- scores: 2 sub-tiles x 2 key tiles, K-dim = DH = 64 ----
        f32x8 s[2][2] = {};
#pragma unroll
        for (int f = 0; f < 2; ++f) {
            const bf16_t* kr0 = kh + (size_t)(kb + row) * DH_;
            const bf16_t* kr1 = kh + (size_t)(kb + 16 + row) * DH_;
            bf16x8 l0 = *(const bf16x8*)(kr0 + f * 32 + 8 * half);
            bf16x8 h0 = *(const bf16x8*)(kr0 + f * 32 + 16 + 8 * half);
            bf16x8 l1 = *(const bf16x8*)(kr1 + f * 32 + 8 * half);
            bf16x8 h1 = *(const bf16x8*)(kr1 + f * 32 + 16 + 8 * half);
            bf16x16 B0, B1;
#pragma unroll
            for (int i = 0; i < 8; ++i) {
                B0[i] = l0[i]; B0[8 + i] = h0[i];
                B1[i] = l1[i]; B1[8 + i] = h1[i];
            }
#pragma unroll
            for (int mi = 0; mi < 2; ++mi) {
                s[mi][0] = __builtin_amdgcn_wmma_f32_16x16x32_bf16(
                    false, QA[mi][f], false, B0, (short)0, s[mi][0], false, false);
                s[mi][1] = __builtin_amdgcn_wmma_f32_16x16x32_bf16(
                    false, QA[mi][f], false, B1, (short)0, s[mi][1], false, false);
            }
        }

        // ---- scale + causal mask + online softmax per sub-tile ----
#pragma unroll
        for (int mi = 0; mi < 2; ++mi) {
            float corr[8];
#pragma unroll
            for (int v = 0; v < 8; ++v) {
                int qrow = q0 + mi * 16 + v + 8 * half;
                float x0 = s[mi][0][v] * 0.125f;
                float x1 = s[mi][1][v] * 0.125f;
                if (kb + row > qrow)      x0 = -1.0e30f;
                if (kb + 16 + row > qrow) x1 = -1.0e30f;

                float mv = fmaxf(x0, x1);
#pragma unroll
                for (int off = 1; off < 16; off <<= 1)
                    mv = fmaxf(mv, __shfl_xor(mv, off, 32));
                float mnew = fmaxf(mrow[mi][v], mv);
                float cr   = __expf(mrow[mi][v] - mnew);
                float p0 = __expf(x0 - mnew);
                float p1 = __expf(x1 - mnew);
                float rs = p0 + p1;
#pragma unroll
                for (int off = 1; off < 16; off <<= 1)
                    rs += __shfl_xor(rs, off, 32);
                lrow[mi][v] = lrow[mi][v] * cr + rs;
                mrow[mi][v] = mnew;
                corr[v] = cr;
                s[mi][0][v] = p0;
                s[mi][1][v] = p1;
            }
#pragma unroll
            for (int t = 0; t < 4; ++t)
#pragma unroll
                for (int v = 0; v < 8; ++v) acc[mi][t][v] *= corr[v];

            // P: C layout -> LDS (same-wave DS ops are in-order)
#pragma unroll
            for (int v = 0; v < 8; ++v) {
                int m = v + 8 * half;
                pl[mi * 512 + m * 32 + row]      = (bf16_t)s[mi][0][v];
                pl[mi * 512 + m * 32 + 16 + row] = (bf16_t)s[mi][1][v];
            }
        }

        // ---- P A-fragments back from LDS ----
        bf16x16 PA[2];
#pragma unroll
        for (int mi = 0; mi < 2; ++mi) {
            bf16x8 lo = *(const bf16x8*)(pl + mi * 512 + row * 32 + 8 * half);
            bf16x8 hi = *(const bf16x8*)(pl + mi * 512 + row * 32 + 16 + 8 * half);
#pragma unroll
            for (int i = 0; i < 8; ++i) { PA[mi][i] = lo[i]; PA[mi][8 + i] = hi[i]; }
        }

        // ---- PV: 4 dh tiles, V B-frag shared by both query sub-tiles ----
#pragma unroll
        for (int t = 0; t < 4; ++t) {
            const bf16_t* vr = vh + (size_t)(t * 16 + row) * S_ + kb;
            bf16x8 lo = *(const bf16x8*)(vr + 8 * half);
            bf16x8 hi = *(const bf16x8*)(vr + 16 + 8 * half);
            bf16x16 VB;
#pragma unroll
            for (int i = 0; i < 8; ++i) { VB[i] = lo[i]; VB[8 + i] = hi[i]; }
#pragma unroll
            for (int mi = 0; mi < 2; ++mi)
                acc[mi][t] = __builtin_amdgcn_wmma_f32_16x16x32_bf16(
                    false, PA[mi], false, VB, (short)0, acc[mi][t], false, false);
        }
    }

    // ---- normalize + store fp32 output [B,S,D] ----
#pragma unroll
    for (int mi = 0; mi < 2; ++mi)
#pragma unroll
        for (int t = 0; t < 4; ++t)
#pragma unroll
            for (int v = 0; v < 8; ++v) {
                int s  = q0 + mi * 16 + v + 8 * half;
                int dh = t * 16 + row;
                out[((size_t)b * S_ + s) * D_ + h * DH_ + dh] =
                    acc[mi][t][v] / lrow[mi][v];
            }
}

// ---------------------------------------------------------------------------
extern "C" void kernel_launch(void* const* d_in, const int* in_sizes, int n_in,
                              void* d_out, int out_size, void* d_ws, size_t ws_size,
                              hipStream_t stream) {
    const float* x  = (const float*)d_in[0];
    const float* Wq = (const float*)d_in[1];
    const float* bq = (const float*)d_in[2];
    const float* Wk = (const float*)d_in[3];
    const float* bk = (const float*)d_in[4];
    const float* Wv = (const float*)d_in[5];
    const float* bv = (const float*)d_in[6];
    float* out = (float*)d_out;

    bf16_t* xb  = (bf16_t*)d_ws;                         // M_*D_
    bf16_t* wtq = xb  + (size_t)M_ * D_;                 // D_*D_
    bf16_t* wtk = wtq + (size_t)D_ * D_;
    bf16_t* wtv = wtk + (size_t)D_ * D_;
    bf16_t* qb  = wtv + (size_t)D_ * D_;                 // B*H*S*DH each
    bf16_t* kb  = qb  + (size_t)B_ * H_ * S_ * DH_;
    bf16_t* vtb = kb  + (size_t)B_ * H_ * S_ * DH_;

    cvt_x_kernel<<<(M_ * D_ / 4 + 255) / 256, 256, 0, stream>>>(x, xb);

    dim3 tb(32, 8);
    dim3 tg(D_ / 32, D_ / 32);
    transpose_w_kernel<<<tg, tb, 0, stream>>>(Wq, wtq);
    transpose_w_kernel<<<tg, tb, 0, stream>>>(Wk, wtk);
    transpose_w_kernel<<<tg, tb, 0, stream>>>(Wv, wtv);

    // 3 matrices * 64 * 16 (64x64) tiles, 8 waves per 256-thread block
    qkv_gemm_kernel<<<(3 * 64 * 16) / 8, 256, 0, stream>>>(
        xb, wtq, wtk, wtv, bq, bk, bv, qb, kb, vtb);

    // B*H*(S/32) waves, 4 waves per 128-thread block
    attn_kernel<<<(B_ * H_ * (S_ / 32)) / 4, 128, 0, stream>>>(qb, kb, vtb, out);
}
